// Attention_4398046511431
// MI455X (gfx1250) — matrix operand order repaired and checked by reading, not compiled, not run on hardware
//
#include <hip/hip_runtime.h>
#include <hip/hip_bf16.h>

typedef __attribute__((ext_vector_type(16))) _Float16 v16h;
typedef __attribute__((ext_vector_type(8)))  float    v8f;
typedef __attribute__((ext_vector_type(8)))  int      v8i;
typedef __attribute__((ext_vector_type(4)))  _Float16 v4h;

#define HIDN   1024
#define HEAD   64
#define BATCH  4
#define SEQ    2048
#define NSPLIT 4
#define KEYS_PER_SPLIT (SEQ / NSPLIT)   // 512

__device__ __forceinline__ v8f wmma16(v16h a, v16h b, v8f c) {
  return __builtin_amdgcn_wmma_f32_16x16x32_f16(
      false, a, false, b, (short)0, c, false, false);
}

__device__ __forceinline__ v8f v8f_zero() {
  v8f r;
#pragma unroll
  for (int i = 0; i < 8; ++i) r[i] = 0.0f;
  return r;
}

// A-matrix 16x32 f16 layout: lanes 0-15 -> K[0..7],K[16..23]; lanes 16-31 -> K[8..15],K[24..31]
__device__ __forceinline__ v16h ld_a16(const _Float16* row, int half) {
  const int4* p = (const int4*)(row + 8 * half);
  int4 lo = p[0];
  int4 hi = p[2];
  v8i t;
  t[0] = lo.x; t[1] = lo.y; t[2] = lo.z; t[3] = lo.w;
  t[4] = hi.x; t[5] = hi.y; t[6] = hi.z; t[7] = hi.w;
  return __builtin_bit_cast(v16h, t);
}

// B-matrix 32x16 f16 layout: lane=column; lanes 0-15 -> K[0..7],K[16..23]; lanes 16-31 -> K[8..15],K[24..31]
__device__ __forceinline__ v16h ld_b16(const _Float16* col, int half) {
  const int4* p = (const int4*)(col + 8 * half);
  int4 lo = p[0];
  int4 hi = p[2];
  v8i t;
  t[0] = lo.x; t[1] = lo.y; t[2] = lo.z; t[3] = lo.w;
  t[4] = hi.x; t[5] = hi.y; t[6] = hi.z; t[7] = hi.w;
  return __builtin_bit_cast(v16h, t);
}

__device__ __forceinline__ int pack_f16x2(float a, float b) {
  unsigned short ua = __builtin_bit_cast(unsigned short, (_Float16)a);
  unsigned short ub = __builtin_bit_cast(unsigned short, (_Float16)b);
  return (int)ua | ((int)ub << 16);
}

// A-operand from f32 source with fused f32->f16 conversion
__device__ __forceinline__ v16h ld_a32cvt(const float* row, int half) {
  v8i t;
#pragma unroll
  for (int v = 0; v < 8; ++v) {
    int koff = ((v < 4) ? 2 * v : 16 + 2 * (v - 4)) + 8 * half;
    float2 f = *(const float2*)(row + koff);
    t[v] = pack_f16x2(f.x, f.y);
  }
  return __builtin_bit_cast(v16h, t);
}

// ---------------- stage 0: weights f32 -> f16 ----------------
__global__ void __launch_bounds__(256) cvt_f32_f16(const float4* __restrict__ src,
                                                   v4h* __restrict__ dst, int n4) {
  int i = blockIdx.x * 256 + threadIdx.x;
  if (i < n4) {
    float4 f = src[i];
    v4h h;
    h[0] = (_Float16)f.x; h[1] = (_Float16)f.y;
    h[2] = (_Float16)f.z; h[3] = (_Float16)f.w;
    dst[i] = h;
  }
}

// ---------------- stage 1: QKV projection, one wave per (proj, batch, seq-tile) ----------------
// Ping-pong pipelined: buffers (a0,b0)/(a1,b1) alternate roles, so next-step loads are
// always in flight under the current 4 WMMAs with zero rotation copies. 1536 waves.
__global__ void __launch_bounds__(256) qkv_kernel(const float* __restrict__ x,
                                                  const _Float16* __restrict__ wh,
                                                  const float* __restrict__ bq,
                                                  const float* __restrict__ bk,
                                                  const float* __restrict__ bv,
                                                  _Float16* __restrict__ qkvh) {
  const int lane = threadIdx.x & 31;
  const int wave = threadIdx.x >> 5;
  const int w    = blockIdx.x * 8 + wave;
  const int proj  = w >> 9;
  const int rem   = w & 511;
  const int batch = rem >> 7;
  const int mt    = rem & 127;
  const int nl    = lane & 15;
  const int half  = lane >> 4;

  const float*    xrow  = x  + ((size_t)batch * SEQ + (size_t)mt * 16 + nl) * HIDN;
  const _Float16* wbase = wh + ((size_t)proj * HEAD + nl) * HIDN;

  v8f acc[4];
#pragma unroll
  for (int t = 0; t < 4; ++t) acc[t] = v8f_zero();

  v16h a0, a1, b0[4], b1[4];
  a0 = ld_a32cvt(xrow, half);
#pragma unroll
  for (int t = 0; t < 4; ++t) b0[t] = ld_b16(wbase + (size_t)(t * 16) * HIDN, half);

  // iterations cover steps kk and kk+32; loads for kk+32 / kk+64 issued before each WMMA group
  for (int kk = 0; kk < HIDN - 64; kk += 64) {       // kk = 0,64,...,896
    a1 = ld_a32cvt(xrow + kk + 32, half);
#pragma unroll
    for (int t = 0; t < 4; ++t) b1[t] = ld_b16(wbase + (size_t)(t * 16) * HIDN + kk + 32, half);
#pragma unroll
    for (int t = 0; t < 4; ++t) acc[t] = wmma16(a0, b0[t], acc[t]);

    a0 = ld_a32cvt(xrow + kk + 64, half);
#pragma unroll
    for (int t = 0; t < 4; ++t) b0[t] = ld_b16(wbase + (size_t)(t * 16) * HIDN + kk + 64, half);
#pragma unroll
    for (int t = 0; t < 4; ++t) acc[t] = wmma16(a1, b1[t], acc[t]);
  }
  // epilogue: steps 960 (in a0/b0) and 992
  a1 = ld_a32cvt(xrow + (HIDN - 32), half);
#pragma unroll
  for (int t = 0; t < 4; ++t) b1[t] = ld_b16(wbase + (size_t)(t * 16) * HIDN + HIDN - 32, half);
#pragma unroll
  for (int t = 0; t < 4; ++t) acc[t] = wmma16(a0, b0[t], acc[t]);
#pragma unroll
  for (int t = 0; t < 4; ++t) acc[t] = wmma16(a1, b1[t], acc[t]);

  // bias (+ q scale), store q,k row-major f16 and v transposed f16
  const float* bias = (proj == 0) ? bq : (proj == 1) ? bk : bv;
#pragma unroll
  for (int t = 0; t < 4; ++t) {
    const int n = t * 16 + nl;
    const float bn = bias[n];
#pragma unroll
    for (int r = 0; r < 8; ++r) {
      const int m = r + 8 * half;
      float val = acc[t][r] + bn;
      if (proj == 0) val *= 0.125f;                 // fold 1/sqrt(64) into q
      if (proj < 2) {
        qkvh[(((size_t)proj * BATCH + batch) * SEQ + (size_t)mt * 16 + m) * HEAD + n] =
            (_Float16)val;
      } else {                                      // v transposed: [batch][64][2048]
        qkvh[((size_t)(2 * BATCH + batch)) * ((size_t)SEQ * HEAD) +
             (size_t)n * SEQ + (size_t)mt * 16 + m] = (_Float16)val;
      }
    }
  }
}

// ---------------- stage 2: flash attention with split-K, one wave per (split, batch, q-tile) ----
__global__ void __launch_bounds__(256) attn_kernel(const _Float16* __restrict__ qkvh,
                                                   float* __restrict__ pacc,
                                                   float* __restrict__ pm,
                                                   float* __restrict__ pl) {
  __shared__ _Float16 lds_p[8 * 512];
  const int lane = threadIdx.x & 31;
  const int wave = threadIdx.x >> 5;
  const int w    = blockIdx.x * 8 + wave;
  const int split = w >> 9;
  const int rem   = w & 511;
  const int batch = rem >> 7;
  const int qt    = rem & 127;
  const int nl    = lane & 15;
  const int half  = lane >> 4;
  _Float16* pbuf = lds_p + wave * 512;

  const size_t plane = (size_t)SEQ * HEAD;
  const _Float16* qh = qkvh + (size_t)batch * plane;
  const _Float16* kh = qkvh + (size_t)(BATCH + batch) * plane;
  const _Float16* vt = qkvh + (size_t)(2 * BATCH + batch) * plane;   // [64][2048]

  const _Float16* qrow = qh + ((size_t)qt * 16 + nl) * HEAD;
  v16h aq[2];
#pragma unroll
  for (int c = 0; c < 2; ++c) aq[c] = ld_a16(qrow + 32 * c, half);

  v8f acc[4];
#pragma unroll
  for (int t = 0; t < 4; ++t) acc[t] = v8f_zero();
  float mrow[8], lrow[8];
#pragma unroll
  for (int r = 0; r < 8; ++r) { mrow[r] = -1e30f; lrow[r] = 0.0f; }

  // one 32-key block: V tiles + next K tiles in flight first, then S-WMMA,
  // softmax, LDS transpose, PV-WMMA. bkc = this block's K tiles, bkn = next block's.
  auto kstep = [&](int j, const v16h* bkc, v16h* bkn) {
    v16h bvt[4];
#pragma unroll
    for (int t4 = 0; t4 < 4; ++t4)
      bvt[t4] = ld_b16(vt + (size_t)(t4 * 16 + nl) * SEQ + j, half);
    bkn[0] = ld_b16(kh + (size_t)(j + 32 + nl) * HEAD, half);
    bkn[1] = ld_b16(kh + (size_t)(j + 32 + nl) * HEAD + 32, half);
    bkn[2] = ld_b16(kh + (size_t)(j + 48 + nl) * HEAD, half);
    bkn[3] = ld_b16(kh + (size_t)(j + 48 + nl) * HEAD + 32, half);

    v8f s0 = v8f_zero(), s1 = v8f_zero();
    s0 = wmma16(aq[0], bkc[0], s0);
    s0 = wmma16(aq[1], bkc[1], s0);
    s1 = wmma16(aq[0], bkc[2], s1);
    s1 = wmma16(aq[1], bkc[3], s1);

#pragma unroll
    for (int r = 0; r < 8; ++r) {
      float t = fmaxf(s0[r], s1[r]);
#pragma unroll
      for (int off = 8; off > 0; off >>= 1)
        t = fmaxf(t, __shfl_xor(t, off, 32));
      const float mnew  = fmaxf(mrow[r], t);
      const float alpha = __expf(mrow[r] - mnew);
      mrow[r] = mnew;
      const float p0 = __expf(s0[r] - mnew);
      const float p1 = __expf(s1[r] - mnew);
      float rs = p0 + p1;
#pragma unroll
      for (int off = 8; off > 0; off >>= 1)
        rs += __shfl_xor(rs, off, 32);
      lrow[r] = lrow[r] * alpha + rs;
#pragma unroll
      for (int t4 = 0; t4 < 4; ++t4) acc[t4][r] *= alpha;
      const int m = r + 8 * half;
      pbuf[m * 32 + nl]      = (_Float16)p0;
      pbuf[m * 32 + nl + 16] = (_Float16)p1;
    }

    v16h ap = ld_a16(pbuf + nl * 32, half);
#pragma unroll
    for (int t4 = 0; t4 < 4; ++t4)
      acc[t4] = wmma16(ap, bvt[t4], acc[t4]);
  };

  const int j0 = split * KEYS_PER_SPLIT;
  const int j1 = j0 + KEYS_PER_SPLIT;

  v16h bkA[4], bkB[4];
  bkA[0] = ld_b16(kh + (size_t)(j0 + nl) * HEAD, half);
  bkA[1] = ld_b16(kh + (size_t)(j0 + nl) * HEAD + 32, half);
  bkA[2] = ld_b16(kh + (size_t)(j0 + 16 + nl) * HEAD, half);
  bkA[3] = ld_b16(kh + (size_t)(j0 + 16 + nl) * HEAD + 32, half);

  // ping-pong: no buffer rotation copies. Look-ahead loads at j1..j1+31 stay
  // inside the resident qkvh workspace (next planes), unused.
  for (int j = j0; j < j1; j += 64) {
    kstep(j,      bkA, bkB);
    kstep(j + 32, bkB, bkA);
  }

  // store unnormalized partials + per-row stats
  const size_t prow_base = ((size_t)split * BATCH + batch) * SEQ + (size_t)qt * 16;
#pragma unroll
  for (int r = 0; r < 8; ++r) {
    const int m = r + 8 * half;
#pragma unroll
    for (int t4 = 0; t4 < 4; ++t4) {
      const int n = t4 * 16 + nl;
      pacc[(prow_base + m) * HEAD + n] = acc[t4][r];
    }
    if (nl == 0) {
      pm[prow_base + m] = mrow[r];
      pl[prow_base + m] = lrow[r];
    }
  }
}

// ---------------- stage 3: combine the NSPLIT partials ----------------
__global__ void __launch_bounds__(256) combine_kernel(const float* __restrict__ pacc,
                                                      const float* __restrict__ pm,
                                                      const float* __restrict__ pl,
                                                      float* __restrict__ out) {
  const int i = blockIdx.x * 256 + threadIdx.x;     // over BATCH*SEQ*HEAD
  if (i >= BATCH * SEQ * HEAD) return;
  const int n     = i & (HEAD - 1);
  const int row   = (i >> 6) & (SEQ - 1);
  const int batch = i >> 17;

  float m = -1e30f;
#pragma unroll
  for (int s = 0; s < NSPLIT; ++s)
    m = fmaxf(m, pm[((size_t)s * BATCH + batch) * SEQ + row]);
  float num = 0.0f, den = 0.0f;
#pragma unroll
  for (int s = 0; s < NSPLIT; ++s) {
    const size_t rb = ((size_t)s * BATCH + batch) * SEQ + row;
    const float e = __expf(pm[rb] - m);
    num += pacc[rb * HEAD + n] * e;
    den += pl[rb] * e;
  }
  out[i] = num / den;
}

extern "C" void kernel_launch(void* const* d_in, const int* in_sizes, int n_in,
                              void* d_out, int out_size, void* d_ws, size_t ws_size,
                              hipStream_t stream) {
  const float* x  = (const float*)d_in[0];
  const float* Wq = (const float*)d_in[1];
  const float* bq = (const float*)d_in[2];
  const float* Wk = (const float*)d_in[3];
  const float* bk = (const float*)d_in[4];
  const float* Wv = (const float*)d_in[5];
  const float* bv = (const float*)d_in[6];
  float* out = (float*)d_out;

  char* ws = (char*)d_ws;
  _Float16* wh   = (_Float16*)ws;                               // 384 KiB
  _Float16* qkvh = (_Float16*)(ws + 393216);                    // 3 MiB
  float*    pacc = (float*)(ws + 393216 + 3145728);             // 8 MiB
  float*    pm   = (float*)(ws + 393216 + 3145728 + 8388608);   // 128 KiB
  float*    pl   = (float*)(ws + 393216 + 3145728 + 8388608 + 131072);

  const int wn4 = (HEAD * HIDN) / 4;
  cvt_f32_f16<<<(wn4 + 255) / 256, 256, 0, stream>>>((const float4*)Wq, (v4h*)wh, wn4);
  cvt_f32_f16<<<(wn4 + 255) / 256, 256, 0, stream>>>((const float4*)Wk, (v4h*)(wh + HEAD * HIDN), wn4);
  cvt_f32_f16<<<(wn4 + 255) / 256, 256, 0, stream>>>((const float4*)Wv, (v4h*)(wh + 2 * HEAD * HIDN), wn4);

  qkv_kernel<<<192, 256, 0, stream>>>(x, wh, bq, bk, bv, qkvh);
  attn_kernel<<<256, 256, 0, stream>>>(qkvh, pacc, pm, pl);
  combine_kernel<<<(BATCH * SEQ * HEAD) / 256, 256, 0, stream>>>(pacc, pm, pl, out);
}
